// HybridDecoder_4123168604877
// MI455X (gfx1250) — matrix operand order repaired and checked by reading, not compiled, not run on hardware
//
#include <hip/hip_runtime.h>

typedef __attribute__((ext_vector_type(4)))  float    v4f;
typedef __attribute__((ext_vector_type(8)))  float    v8f;
typedef __attribute__((ext_vector_type(4)))  _Float16 v4h;
typedef __attribute__((ext_vector_type(8)))  _Float16 v8h;
typedef __attribute__((ext_vector_type(16))) _Float16 v16h;

#define DIM    128
#define HIDDEN 32

static __device__ __forceinline__ v16h cat8(v8h lo, v8h hi) {
  return __builtin_shufflevector(lo, hi, 0,1,2,3,4,5,6,7,8,9,10,11,12,13,14,15);
}

// ---------------------------------------------------------------------------
// Kernel 1: convert z (f32, row-major N x 128) -> zh (f16), 8 elems/thread
// ---------------------------------------------------------------------------
__global__ void zh_convert_kernel(const float* __restrict__ z,
                                  _Float16* __restrict__ zh, int n8) {
  int i = blockIdx.x * blockDim.x + threadIdx.x;
  if (i >= n8) return;
  v4f a = *(const v4f*)(z + (long)i * 8);
  v4f b = *(const v4f*)(z + (long)i * 8 + 4);
  v8h h;
  h[0] = (_Float16)a[0]; h[1] = (_Float16)a[1];
  h[2] = (_Float16)a[2]; h[3] = (_Float16)a[3];
  h[4] = (_Float16)b[0]; h[5] = (_Float16)b[1];
  h[6] = (_Float16)b[2]; h[7] = (_Float16)b[3];
  *(v8h*)(zh + (long)i * 8) = h;
}

// ---------------------------------------------------------------------------
// Kernel 2: pack B-matrix fragments in exact WMMA lane layout.
// B (32x16 f16 chunk): lanes 0-15 hold N=lane, K=0..7,16..23 ; lanes 16-31 hold
// N=lane-16, K=8..15,24..31 ; 16 K per lane -> contiguous 32B per lane.
// bp1: Wcat = [W | w1a | w1b]  (128 x 192) -> [nt(12)][c(4)][lane(32)][m(16)]
// bp2: w1c (128 x 32)          -> [nt(2)][c(4)][lane(32)][m(16)]
// ---------------------------------------------------------------------------
__global__ void pack_kernel(const float* __restrict__ W,
                            const float* __restrict__ w1,
                            _Float16* __restrict__ bp1,
                            _Float16* __restrict__ bp2) {
  int tid = blockIdx.x * blockDim.x + threadIdx.x;
  const int N1 = 12 * 4 * 32 * 16;
  const int N2 = 2 * 4 * 32 * 16;
  if (tid < N1) {
    int m  = tid & 15;
    int l  = (tid >> 4) & 31;
    int c  = (tid >> 9) & 3;
    int nt = tid >> 11;
    int n  = nt * 16 + (l & 15);
    int k  = c * 32 + ((l < 16) ? 0 : 8) + m + ((m < 8) ? 0 : 8);
    float v;
    if (n < 128)      v = W[k * 128 + n];                 // W block
    else if (n < 160) v = w1[k * 32 + (n - 128)];         // w1a block
    else              v = w1[(128 + k) * 32 + (n - 160)]; // w1b block
    bp1[tid] = (_Float16)v;
  } else if (tid < N1 + N2) {
    int idx = tid - N1;
    int m  = idx & 15;
    int l  = (idx >> 4) & 31;
    int c  = (idx >> 9) & 3;
    int nt = idx >> 11;
    int n  = nt * 16 + (l & 15);
    int k  = c * 32 + ((l < 16) ? 0 : 8) + m + ((m < 8) ? 0 : 8);
    bp2[idx] = (_Float16)w1[(256 + k) * 32 + n];          // w1c block
  }
}

// ---------------------------------------------------------------------------
// Kernel 3: node precompute GEMM  P = zh @ [W | w1a | w1b]   (N x 192)
//   cols   0..127 -> ZW  (f32, N x 128)
//   cols 128..191 -> AB  (f32, N x 64: A=z@w1a in 0..31, B=z@w1b in 32..63)
// One wave per 16-row M-tile: A fragments (4 x v16h) loaded ONCE, reused for
// all 12 N-tiles -> 48 WMMAs per wave, 12x less A traffic. ZW/AB split and
// tail guard resolved uniformly (no divergent stores).
// ---------------------------------------------------------------------------
__global__ void __launch_bounds__(128)
node_gemm_kernel(const _Float16* __restrict__ zh,
                 const _Float16* __restrict__ bp1,
                 float* __restrict__ ZW,
                 float* __restrict__ AB,
                 int nNodes, int tilesM) {
  int wid  = (blockIdx.x * blockDim.x + threadIdx.x) >> 5;
  int lane = threadIdx.x & 31;
  if (wid >= tilesM) return;
  int m0 = wid * 16;

  int row = m0 + (lane & 15);
  if (row >= nNodes) row = nNodes - 1;
  const _Float16* zrow = zh + (long)row * DIM;
  int g = (lane < 16) ? 0 : 8;   // A layout: lanes>=16 hold K+8 / K+24

  // Load all 4 A fragments once (K = 0..127)
  v16h afrag[4];
#pragma unroll
  for (int c = 0; c < 4; ++c) {
    v8h alo = *(const v8h*)(zrow + c * 32 + g);
    v8h ahi = *(const v8h*)(zrow + c * 32 + g + 16);
    afrag[c] = cat8(alo, ahi);
  }

  // C/D layout: VGPR r -> row r (lanes 0-15) / row r+8 (lanes 16-31), col lane%16
  int  col  = lane & 15;
  int  gm   = (lane < 16) ? 0 : 8;
  bool full = (m0 + 16 <= nNodes);          // uniform: interior tile
  float* zwBase = ZW + (long)(m0 + gm) * DIM + col;
  float* abBase = AB + (long)(m0 + gm) * 64 + col;

#pragma unroll
  for (int nt = 0; nt < 12; ++nt) {
    v8f acc = {};
#pragma unroll
    for (int c = 0; c < 4; ++c) {
      v16h b = *(const v16h*)(bp1 + ((nt * 4 + c) * 32 + lane) * 16);
      acc = __builtin_amdgcn_wmma_f32_16x16x32_f16(false, afrag[c], false, b,
                                                   (short)0, acc, false, false);
    }
    if (nt < 8) {                 // -> ZW, cols nt*16 .. nt*16+15
      float* p = zwBase + nt * 16;
      if (full) {
#pragma unroll
        for (int r = 0; r < 8; ++r) p[(long)r * DIM] = acc[r];
      } else {
#pragma unroll
        for (int r = 0; r < 8; ++r)
          if (m0 + gm + r < nNodes) p[(long)r * DIM] = acc[r];
      }
    } else {                      // -> AB, cols (nt-8)*16 .. +15
      float* p = abBase + (nt - 8) * 16;
      if (full) {
#pragma unroll
        for (int r = 0; r < 8; ++r) p[(long)r * 64] = acc[r];
      } else {
#pragma unroll
        for (int r = 0; r < 8; ++r)
          if (m0 + gm + r < nNodes) p[(long)r * 64] = acc[r];
      }
    }
  }
}

// ---------------------------------------------------------------------------
// Kernel 4: per-edge kernel, one wave per 16 edges.
//   bilinear = dot(ZW[src], z[dst]) in f32
//   mlp      = sum_j relu( (prod@w1c)_j + A[src][j] + B[dst][j] + b1_j ) * w2_j
// prod tile (16 x 128 f16) staged in LDS (row stride 272B to spread banks),
// consumed as WMMA A fragments; w1c consumed as prepacked B fragments.
// ---------------------------------------------------------------------------
__global__ void __launch_bounds__(128)
edge_kernel(const float* __restrict__ z,
            const long long* __restrict__ ei,
            const float* __restrict__ ZW,
            const float* __restrict__ AB,
            const _Float16* __restrict__ bp2,
            const float* __restrict__ b1,
            const float* __restrict__ w2,
            const float* __restrict__ bias,
            const float* __restrict__ b2,
            float* __restrict__ out,
            int E, int tilesE) {
  // per-wave LDS: 32 ints (src/dst) + 16 floats (bilinear) + pad + 16x272B prod
  __shared__ __align__(16) unsigned char smem[4 * 4608];
  int wv   = threadIdx.x >> 5;
  int lane = threadIdx.x & 31;
  int tile = blockIdx.x * 4 + wv;
  if (tile >= tilesE) return;

  unsigned char* base = smem + wv * 4608;
  int*      sd   = (int*)base;              // [0..15]=src, [16..31]=dst
  float*    bil  = (float*)(base + 128);
  _Float16* prod = (_Float16*)(base + 256); // 16 rows, 136 halves (272B) stride

  if (lane < 16) {
    long e = (long)tile * 16 + lane;
    if (e >= E) e = E - 1;
    sd[lane]      = (int)ei[e];
    sd[16 + lane] = (int)ei[(long)E + e];
  }
  asm volatile("s_wait_dscnt 0x0" ::: "memory");

  int i = lane & 15, h = lane >> 4;   // edge-in-tile, K-half (0: k<64, 1: k>=64)
  int si = sd[i], di = sd[16 + i];
  const float* zwrow = ZW + (long)si * DIM + h * 64;
  const float* zsrow = z  + (long)si * DIM + h * 64;
  const float* zdrow = z  + (long)di * DIM + h * 64;
  _Float16*    prow  = prod + i * 136 + h * 64;

  float dacc = 0.f;
#pragma unroll
  for (int j = 0; j < 16; ++j) {
    v4f a = *(const v4f*)(zwrow + 4 * j);
    v4f s = *(const v4f*)(zsrow + 4 * j);
    v4f d = *(const v4f*)(zdrow + 4 * j);
    dacc += a[0] * d[0] + a[1] * d[1] + a[2] * d[2] + a[3] * d[3];
    v4f p = s * d;
    v4h ph;
    ph[0] = (_Float16)p[0]; ph[1] = (_Float16)p[1];
    ph[2] = (_Float16)p[2]; ph[3] = (_Float16)p[3];
    *(v4h*)(prow + 4 * j) = ph;
  }
  dacc += __shfl_xor(dacc, 16, 32);   // combine the two K-halves
  if (lane < 16) bil[lane] = dacc;
  asm volatile("s_wait_dscnt 0x0" ::: "memory");

  // 8 WMMAs: prod (16x128) @ w1c (128x32), N split into two 16-wide tiles
  int g16 = (lane < 16) ? 0 : 8;
  v8f c0 = {}, c1 = {};
#pragma unroll
  for (int c = 0; c < 4; ++c) {
    const _Float16* ar = prod + i * 136 + c * 32 + g16;
    v16h a   = cat8(*(const v8h*)ar, *(const v8h*)(ar + 16));
    v16h b0  = *(const v16h*)(bp2 + ((0 * 4 + c) * 32 + lane) * 16);
    v16h b1f = *(const v16h*)(bp2 + ((1 * 4 + c) * 32 + lane) * 16);
    c0 = __builtin_amdgcn_wmma_f32_16x16x32_f16(false, a, false, b0,
                                                (short)0, c0, false, false);
    c1 = __builtin_amdgcn_wmma_f32_16x16x32_f16(false, a, false, b1f,
                                                (short)0, c1, false, false);
  }

  // epilogue: relu + w2 dot, half-wave reductions
  int j0 = lane & 15, j1 = j0 + 16;
  float b1j0 = b1[j0], b1j1 = b1[j1];
  float w2j0 = w2[j0], w2j1 = w2[j1];
  float bsum = bias[0] + b2[0];
  int gm = (lane < 16) ? 0 : 8;
#pragma unroll
  for (int r = 0; r < 8; ++r) {
    int M = r + gm;                 // edge-in-tile owned by this half-wave
    long e = (long)tile * 16 + M;
    int sM = sd[M], dM = sd[16 + M];
    float h0 = c0[r] + AB[(long)sM * 64 + j0] + AB[(long)dM * 64 + 32 + j0] + b1j0;
    float h1 = c1[r] + AB[(long)sM * 64 + j1] + AB[(long)dM * 64 + 32 + j1] + b1j1;
    float v = fmaxf(h0, 0.f) * w2j0 + fmaxf(h1, 0.f) * w2j1;
    v += __shfl_xor(v, 1, 32);
    v += __shfl_xor(v, 2, 32);
    v += __shfl_xor(v, 4, 32);
    v += __shfl_xor(v, 8, 32);      // lanes with j0==0 hold the edge sum
    if (j0 == 0 && e < E) out[e] = v + bil[M] + bsum;
  }
}

// ---------------------------------------------------------------------------
extern "C" void kernel_launch(void* const* d_in, const int* in_sizes, int n_in,
                              void* d_out, int out_size, void* d_ws, size_t ws_size,
                              hipStream_t stream) {
  const float*     z    = (const float*)d_in[0];
  const long long* ei   = (const long long*)d_in[1];   // int64 edge_index (2,E)
  const float*     W    = (const float*)d_in[2];
  const float*     bias = (const float*)d_in[3];
  const float*     w1   = (const float*)d_in[4];
  const float*     b1   = (const float*)d_in[5];
  const float*     w2   = (const float*)d_in[6];
  const float*     b2   = (const float*)d_in[7];
  float*           out  = (float*)d_out;

  int N = in_sizes[0] / DIM;   // 100000
  int E = in_sizes[1] / 2;     // 500000

  // workspace layout (256B aligned slabs)
  char* ws = (char*)d_ws;
  size_t off = 0;
  auto take = [&](size_t bytes) {
    char* p = ws + off;
    off += (bytes + 255) & ~(size_t)255;
    return p;
  };
  _Float16* zh  = (_Float16*)take((size_t)N * DIM * sizeof(_Float16)); // 25.6MB
  float*    ZW  = (float*)   take((size_t)N * DIM * sizeof(float));    // 51.2MB
  float*    AB  = (float*)   take((size_t)N * 64  * sizeof(float));    // 25.6MB
  _Float16* bp1 = (_Float16*)take((size_t)12 * 4 * 32 * 16 * sizeof(_Float16));
  _Float16* bp2 = (_Float16*)take((size_t)2  * 4 * 32 * 16 * sizeof(_Float16));

  // 1) z -> f16
  int n8 = (N * DIM) / 8;
  zh_convert_kernel<<<(n8 + 255) / 256, 256, 0, stream>>>(z, zh, n8);

  // 2) pack WMMA B fragments
  int npack = 12 * 4 * 32 * 16 + 2 * 4 * 32 * 16;
  pack_kernel<<<(npack + 255) / 256, 256, 0, stream>>>(W, w1, bp1, bp2);

  // 3) node precompute: ZW, A, B  (one wave per 16-row tile, A reused 12x)
  int tilesM = (N + 15) / 16;
  node_gemm_kernel<<<(tilesM * 32 + 127) / 128, 128, 0, stream>>>(
      zh, bp1, ZW, AB, N, tilesM);

  // 4) edge kernel: one wave per 16 edges, 4 waves per block
  int tilesE = (E + 15) / 16;
  edge_kernel<<<(tilesE + 3) / 4, 128, 0, stream>>>(
      z, ei, ZW, AB, bp2, b1, w2, bias, b2, out, E, tilesE);
}
